// BahdanauAttention_40699110097407
// MI455X (gfx1250) — matrix operand order repaired and checked by reading, not compiled, not run on hardware
//
#include <hip/hip_runtime.h>
#include <hip/hip_bf16.h>

#define BB 32
#define SS 2048
#define DD 1024
#define UU 1024
#define MT 64   // rows of enc per block in main kernel

typedef __attribute__((ext_vector_type(4)))  float        f32x4;
typedef __attribute__((ext_vector_type(8)))  float        f32x8;
typedef __attribute__((ext_vector_type(4)))  unsigned int u32x4;
typedef __attribute__((ext_vector_type(16))) __bf16       bf16x16;

union Frag {
    u32x4   q[2];
    bf16x16 v;
};

__device__ __forceinline__ unsigned short f2bf(float f) {
    unsigned int u = __float_as_uint(f);
    unsigned int r = u + 0x7FFFu + ((u >> 16) & 1u);   // round-to-nearest-even
    return (unsigned short)(r >> 16);
}

// Native gfx1250 V_TANH_F32 if the toolchain exposes it; libm fallback otherwise.
__device__ __forceinline__ float fast_tanh(float x) {
#if __has_builtin(__builtin_amdgcn_tanhf)
    return __builtin_amdgcn_tanhf(x);
#elif __has_builtin(__builtin_amdgcn_tanh_f32)
    return __builtin_amdgcn_tanh_f32(x);
#else
    return tanhf(x);
#endif
}

// ---------------------------------------------------------------------------
// Kernel 1: WsT[u][d] = bf16(Ws_w[d][u])   (1024x1024), 32x32 tiles via LDS
// ---------------------------------------------------------------------------
__global__ __launch_bounds__(256) void prep_wst(const float* __restrict__ Ws,
                                                unsigned short* __restrict__ WsT) {
    __shared__ float t[32][33];
    int bx = blockIdx.x;            // d tile
    int by = blockIdx.y;            // u tile
    int tx = threadIdx.x & 31;
    int ty = threadIdx.x >> 5;      // 0..7
#pragma unroll
    for (int k = 0; k < 4; ++k) {
        int d = bx * 32 + ty + k * 8;
        t[ty + k * 8][tx] = Ws[(size_t)d * UU + by * 32 + tx];
    }
    __syncthreads();
#pragma unroll
    for (int k = 0; k < 4; ++k) {
        int u = by * 32 + ty + k * 8;
        WsT[(size_t)u * DD + bx * 32 + tx] = f2bf(t[tx][ty + k * 8]);
    }
}

// ---------------------------------------------------------------------------
// Kernel 2: bias2[b][u] = dec_hidden[b]·Wh_w[:,u] + Wh_b[u] + Ws_b[u] + Wc_b[u]
// grid 4 blocks x 256 threads, one u per thread, all 32 b in registers
// ---------------------------------------------------------------------------
__global__ __launch_bounds__(256) void prep_bias(const float* __restrict__ dec,
                                                 const float* __restrict__ Whw,
                                                 const float* __restrict__ Whb,
                                                 const float* __restrict__ Wsb,
                                                 const float* __restrict__ Wcb,
                                                 float* __restrict__ bias2) {
    __shared__ float decS[32 * 256];
    int u = blockIdx.x * 256 + threadIdx.x;
    float acc[BB];
#pragma unroll
    for (int b = 0; b < BB; ++b) acc[b] = 0.f;

    for (int d0 = 0; d0 < DD; d0 += 256) {
        for (int i = threadIdx.x; i < 32 * 256; i += 256) {
            int b = i >> 8, dd = i & 255;
            decS[i] = dec[b * DD + d0 + dd];
        }
        __syncthreads();
        for (int dd = 0; dd < 256; ++dd) {
            float wv = Whw[(size_t)(d0 + dd) * UU + u];
#pragma unroll
            for (int b = 0; b < BB; ++b) acc[b] += decS[b * 256 + dd] * wv;
        }
        __syncthreads();
    }
    float bias = Whb[u] + Wsb[u] + Wcb[u];
    for (int b = 0; b < BB; ++b) bias2[b * UU + u] = acc[b] + bias;
}

// ---------------------------------------------------------------------------
// Kernel 3 (main): fused  score[b][s] = sum_u V_w[u]*tanh(enc·Ws + bias2 + cov*Wc)
// One block = 64 rows of enc (bf16 in LDS), 8 waves of WMMA over all 1024 cols.
// ---------------------------------------------------------------------------
__global__ __launch_bounds__(256) void attn_scores(const float* __restrict__ enc,
                                                   const float* __restrict__ prevcov,
                                                   const unsigned short* __restrict__ WsT,
                                                   const float* __restrict__ bias2,
                                                   const float* __restrict__ wc,
                                                   const float* __restrict__ vw,
                                                   float* __restrict__ score) {
    __shared__ unsigned short Abuf[MT * DD];  // 128 KB bf16 slab of enc
    __shared__ float covS[MT];
    __shared__ float part[8][16];             // per-wave row partial scores

    const int tid = threadIdx.x;
    const int blk = blockIdx.x;               // 0..1023
    const int b   = blk / (SS / MT);
    const int s0  = (blk % (SS / MT)) * MT;

    // ---- stage A slab: f32 -> bf16 into LDS ----
    const float* encRow = enc + ((size_t)b * SS + s0) * DD;
    for (int i = tid; i < (MT * DD) / 4; i += 256) {
        f32x4 f = ((const f32x4*)encRow)[i];
        unsigned int lo = (unsigned int)f2bf(f[0]) | ((unsigned int)f2bf(f[1]) << 16);
        unsigned int hi = (unsigned int)f2bf(f[2]) | ((unsigned int)f2bf(f[3]) << 16);
        unsigned int* q = (unsigned int*)(Abuf + i * 4);
        q[0] = lo; q[1] = hi;
    }
    if (tid < MT) covS[tid] = prevcov[b * SS + s0 + tid];
    __syncthreads();

    const int w       = tid >> 5;          // wave 0..7
    const int lane    = tid & 31;
    const int mi      = w & 3;             // M subtile (16 rows)
    const int ncol0   = (w >> 2) * 16;     // column group offset 0/16
    const int halfSel = lane >> 4;         // lane half per ISA fragment layout
    const int l16     = lane & 15;

    const unsigned short* aBase = Abuf + (mi * 16 + l16) * DD + halfSel * 8;
    const int rowBase = mi * 16 + halfSel * 8;

    float sacc[8];
#pragma unroll
    for (int v = 0; v < 8; ++v) sacc[v] = 0.f;

    for (int nt = 0; nt < UU / 32; ++nt) {
        const int col = ncol0 + nt * 32 + l16;                    // this lane's column
        const unsigned short* bBase = WsT + (size_t)col * DD + halfSel * 8;

        f32x8 acc = {};
#pragma unroll 4
        for (int kk = 0; kk < DD; kk += 32) {
            Frag a, bf;
            // A 16x32 bf16: lane<16 -> K {kk..kk+7, kk+16..kk+23}; lane>=16 -> +8
            a.q[0] = *(const u32x4*)(aBase + kk);
            a.q[1] = *(const u32x4*)(aBase + kk + 16);
            // B 32x16 bf16: lane holds column `col`, K {kk..kk+7, kk+16..kk+23} (+8 for lane>=16)
            const u32x4* bp = (const u32x4*)(bBase + kk);
            bf.q[0] = bp[0];
            bf.q[1] = bp[2];
            acc = __builtin_amdgcn_wmma_f32_16x16x32_bf16(false, a.v, false, bf.v,
                                                          (short)0, acc, false, false);
        }
        // epilogue: +bias2 +cov*Wc, native tanh, dot with V_w
        const float bcol = bias2[b * UU + col];
        const float wcc  = wc[col];
        const float vwc  = vw[col];
#pragma unroll
        for (int v = 0; v < 8; ++v) {
            float feat = acc[v] + bcol + covS[rowBase + v] * wcc;
            sacc[v] += fast_tanh(feat) * vwc;
        }
    }

    // reduce across the 16 lanes of each half (columns) -> per-row sums
#pragma unroll
    for (int v = 0; v < 8; ++v) {
        float x = sacc[v];
        x += __shfl_xor(x, 1);
        x += __shfl_xor(x, 2);
        x += __shfl_xor(x, 4);
        x += __shfl_xor(x, 8);
        if (l16 == 0) part[w][halfSel * 8 + v] = x;
    }
    __syncthreads();
    if (tid < MT) {
        int m2 = tid >> 4, r = tid & 15;
        score[b * SS + s0 + tid] = part[m2][r] + part[m2 + 4][r];
    }
}

// ---------------------------------------------------------------------------
// Kernel 4: softmax over S, attention weights + coverage outputs
// ---------------------------------------------------------------------------
__global__ __launch_bounds__(256) void softmax_cov(const float* __restrict__ score,
                                                   const float* __restrict__ prevcov,
                                                   float* __restrict__ attn_out,
                                                   float* __restrict__ cov_out,
                                                   float* __restrict__ w_ws) {
    __shared__ float red[256];
    const int b = blockIdx.x, tid = threadIdx.x;
    const float* sc = score + b * SS;

    float m = -3.4e38f;
    for (int s = tid; s < SS; s += 256) m = fmaxf(m, sc[s]);
    red[tid] = m; __syncthreads();
    for (int st = 128; st > 0; st >>= 1) {
        if (tid < st) red[tid] = fmaxf(red[tid], red[tid + st]);
        __syncthreads();
    }
    m = red[0]; __syncthreads();

    float sum = 0.f;
    for (int s = tid; s < SS; s += 256) sum += __expf(sc[s] - m);
    red[tid] = sum; __syncthreads();
    for (int st = 128; st > 0; st >>= 1) {
        if (tid < st) red[tid] += red[tid + st];
        __syncthreads();
    }
    const float inv = 1.0f / red[0];

    for (int s = tid; s < SS; s += 256) {
        float wv = __expf(sc[s] - m) * inv;
        attn_out[b * SS + s] = wv;
        cov_out[b * SS + s]  = wv + prevcov[b * SS + s];
        w_ws[b * SS + s]     = wv;
    }
}

// ---------------------------------------------------------------------------
// Kernel 5: context[b][d] = sum_s w[b][s] * enc[b][s][d]   (deterministic)
// grid (32, 4): each block owns 256 d-columns, full S loop
// ---------------------------------------------------------------------------
__global__ __launch_bounds__(256) void context_k(const float* __restrict__ enc,
                                                 const float* __restrict__ w_ws,
                                                 float* __restrict__ ctx) {
    __shared__ float wS[SS];
    const int b = blockIdx.x;
    const int d = blockIdx.y * 256 + threadIdx.x;
    for (int s = threadIdx.x; s < SS; s += 256) wS[s] = w_ws[b * SS + s];
    __syncthreads();
    float acc = 0.f;
    const float* e = enc + (size_t)b * SS * DD + d;
#pragma unroll 4
    for (int s = 0; s < SS; ++s) acc += wS[s] * e[(size_t)s * DD];
    ctx[b * DD + d] = acc;
}

// ---------------------------------------------------------------------------
extern "C" void kernel_launch(void* const* d_in, const int* in_sizes, int n_in,
                              void* d_out, int out_size, void* d_ws, size_t ws_size,
                              hipStream_t stream) {
    const float* dec  = (const float*)d_in[0];
    const float* enc  = (const float*)d_in[1];
    // d_in[2] enc_pad_mask (unused), d_in[3] use_coverage (True path)
    const float* pcov = (const float*)d_in[4];
    const float* Wsw  = (const float*)d_in[5];
    const float* Wsb  = (const float*)d_in[6];
    const float* Whw  = (const float*)d_in[7];
    const float* Whb  = (const float*)d_in[8];
    const float* Wcw  = (const float*)d_in[9];
    const float* Wcb  = (const float*)d_in[10];
    const float* Vw   = (const float*)d_in[11];
    // d_in[12] V_b: softmax-invariant shift, intentionally dropped

    float* ctx_out  = (float*)d_out;                 // [32,1024]
    float* attn_out = ctx_out + BB * DD;             // [32,2048]
    float* cov_out  = attn_out + BB * SS;            // [32,2048]

    char* ws = (char*)d_ws;
    unsigned short* WsT = (unsigned short*)ws;                       // 2 MB bf16
    float* bias2  = (float*)(ws + (size_t)UU * DD * 2);              // 128 KB
    float* scores = (float*)(ws + (size_t)UU * DD * 2 + BB * UU * 4);        // 256 KB
    float* w_ws   = (float*)(ws + (size_t)UU * DD * 2 + BB * UU * 4 + BB * SS * 4); // 256 KB

    prep_wst<<<dim3(DD / 32, UU / 32), 256, 0, stream>>>(Wsw, WsT);
    prep_bias<<<UU / 256, 256, 0, stream>>>(dec, Whw, Whb, Wsb, Wcb, bias2);
    attn_scores<<<(BB * SS) / MT, 256, 0, stream>>>(enc, pcov, WsT, bias2, Wcw, Vw, scores);
    softmax_cov<<<BB, 256, 0, stream>>>(scores, pcov, attn_out, cov_out, w_ws);
    context_k<<<dim3(BB, DD / 256), 256, 0, stream>>>(enc, w_ws, ctx_out);
}
